// GCN_scheduling_64957085384788
// MI455X (gfx1250) — matrix operand order repaired and checked by reading, not compiled, not run on hardware
//
#include <hip/hip_runtime.h>
#include <math.h>

typedef __attribute__((ext_vector_type(16))) __bf16 v16bf;
typedef __attribute__((ext_vector_type(8)))  __bf16 v8bf;
typedef __attribute__((ext_vector_type(8)))  float  v8f;

#define N_NODES 30000
#define N_EDGES 300000
#define ET_TOTAL (N_EDGES + N_NODES)
#define HEADS 8

// ---------------- helpers ----------------
__device__ __forceinline__ void edge_sd(const int* __restrict__ ei, int e, int& s, int& d) {
  if (e < N_EDGES) { s = ei[e]; d = ei[N_EDGES + e]; }
  else { s = e - N_EDGES; d = s; }  // self loops appended
}
// monotonic uint encoding of float for atomicMax-based segment max
__device__ __forceinline__ unsigned fenc(float f) {
  unsigned u = __float_as_uint(f);
  return (u & 0x80000000u) ? ~u : (u | 0x80000000u);
}
__device__ __forceinline__ float fdec(unsigned u) {
  unsigned b = (u & 0x80000000u) ? (u & 0x7FFFFFFFu) : ~u;
  return __uint_as_float(b);
}

// ---------------- f32 -> bf16 staging (zero-padded K) ----------------
__global__ void k_convA(const float* __restrict__ src, unsigned short* __restrict__ dst_,
                        int K, int K_pad) {
  __bf16* dst = (__bf16*)dst_;
  long long i = (long long)blockIdx.x * blockDim.x + threadIdx.x;
  long long total = (long long)N_NODES * K_pad;
  if (i >= total) return;
  int k = (int)(i % K_pad);
  long long n = i / K_pad;
  dst[i] = (k < K) ? (__bf16)src[n * K + k] : (__bf16)0.0f;
}

// W [K, Nout] f32 -> Bt [Nout, K_pad] bf16 (transposed, zero-padded)
__global__ void k_convB(const float* __restrict__ W, unsigned short* __restrict__ dst_,
                        int K, int K_pad, int Nout) {
  __bf16* dst = (__bf16*)dst_;
  int i = blockIdx.x * blockDim.x + threadIdx.x;
  int total = Nout * K_pad;
  if (i >= total) return;
  int k = i % K_pad;
  int n = i / K_pad;
  dst[i] = (k < K) ? (__bf16)W[(long long)k * Nout + n] : (__bf16)0.0f;
}

// ---------------- WMMA GEMM: C[M,Nout] = A[M,K_pad] * Bt[Nout,K_pad]^T ----------------
__global__ void k_gemm_wmma(const unsigned short* __restrict__ A_, const unsigned short* __restrict__ B_,
                            float* __restrict__ C, int K_pad, int Nout) {
  const __bf16* A  = (const __bf16*)A_;
  const __bf16* Bt = (const __bf16*)B_;
  const int ntilesN = Nout >> 4;
  const int wave = threadIdx.x >> 5;
  const int lane = threadIdx.x & 31;
  int tile = blockIdx.x * (blockDim.x >> 5) + wave;
  int T = (N_NODES >> 4) * ntilesN;
  if (tile >= T) return;                       // wave-uniform: EXEC stays all-ones
  int mt = tile / ntilesN, nt = tile % ntilesN;
  int half = lane >> 4;
  int lrow = lane & 15;
  const __bf16* Arow = A  + (size_t)(mt * 16 + lrow) * K_pad;  // A frag: lane row M=lrow
  const __bf16* Brow = Bt + (size_t)(nt * 16 + lrow) * K_pad;  // B frag: lane col N=lrow
  v8f acc = {};
  for (int kt = 0; kt < K_pad; kt += 32) {
    // A layout: lane half h holds K = [16h..16h+16) contiguous (matches B)
    v16bf af = *(const v16bf*)(Arow + kt + 16 * half);
    // B layout: lane half h holds K = [16h..16h+16) contiguous (matches A)
    v16bf bfr = *(const v16bf*)(Brow + kt + 16 * half);
    acc = __builtin_amdgcn_wmma_f32_16x16x32_bf16(false, af, false, bfr, (short)0, acc,
                                                  false, false);
  }
  // C/D layout: N = lane&15, M = 8*half + r
  float* Crow = C + (size_t)(mt * 16 + 8 * half) * Nout + nt * 16 + lrow;
#pragma unroll
  for (int r = 0; r < 8; ++r) Crow[(size_t)r * Nout] = acc[r];
}

// ---------------- per-node attention logits ----------------
__global__ void k_node_att(const float* __restrict__ xh, const float* __restrict__ a_src,
                           const float* __restrict__ a_dst, float* __restrict__ als,
                           float* __restrict__ ald, int F) {
  int i = blockIdx.x * blockDim.x + threadIdx.x;
  if (i >= N_NODES * HEADS) return;
  int h = i & 7;
  int n = i >> 3;
  const float* row = xh + (size_t)n * HEADS * F + (size_t)h * F;
  const float* as = a_src + h * F;
  const float* ad = a_dst + h * F;
  float s = 0.f, d = 0.f;
  for (int f = 0; f < F; ++f) { float v = row[f]; s += v * as[f]; d += v * ad[f]; }
  als[i] = s; ald[i] = d;
}

// ---------------- edge logits + segment max ----------------
__global__ void k_edge_logit(const int* __restrict__ ei, const float* __restrict__ als,
                             const float* __restrict__ ald, float* __restrict__ ebuf,
                             unsigned* __restrict__ segmax) {
  long long i = (long long)blockIdx.x * blockDim.x + threadIdx.x;
  if (i >= (long long)ET_TOTAL * HEADS) return;
  int h = (int)(i & 7);
  int e = (int)(i >> 3);
  int s, d; edge_sd(ei, e, s, d);
  float v = als[s * HEADS + h] + ald[d * HEADS + h];
  v = v > 0.f ? v : 0.2f * v;                 // leaky_relu(0.2)
  ebuf[i] = v;
  atomicMax(&segmax[d * HEADS + h], fenc(v));
}

// ---------------- exp + segment sum ----------------
__global__ void k_edge_exp(const int* __restrict__ ei, const unsigned* __restrict__ segmax,
                           float* __restrict__ ebuf, float* __restrict__ denom) {
  long long i = (long long)blockIdx.x * blockDim.x + threadIdx.x;
  if (i >= (long long)ET_TOTAL * HEADS) return;
  int h = (int)(i & 7);
  int e = (int)(i >> 3);
  int s, d; edge_sd(ei, e, s, d);
  float m = fdec(segmax[d * HEADS + h]);
  float v = __expf(ebuf[i] - m);
  ebuf[i] = v;
  atomicAdd(&denom[d * HEADS + h], v);
}

// ---------------- weighted message aggregation (one wave32 per edge) ----------------
__global__ void k_aggregate(const int* __restrict__ ei, const float* __restrict__ xh,
                            const float* __restrict__ ebuf, const float* __restrict__ denom,
                            float* __restrict__ out, int HF, int F) {
  int e = blockIdx.x * (blockDim.x >> 5) + (threadIdx.x >> 5);
  if (e >= ET_TOTAL) return;
  int lane = threadIdx.x & 31;
  int s, d; edge_sd(ei, e, s, d);
  const float* xrow = xh + (size_t)s * HF;
  float* orow = out + (size_t)d * HF;
  for (int g = 0; g < HF; g += 128) {
    int c = g + lane * 4;                     // float4 stays within one head (F>=16, 4|F)
    int h = c / F;
    float alpha = ebuf[(size_t)e * HEADS + h] / (denom[d * HEADS + h] + 1e-16f);
    float4 xv = *(const float4*)(xrow + c);
    atomicAdd(orow + c + 0, xv.x * alpha);
    atomicAdd(orow + c + 1, xv.y * alpha);
    atomicAdd(orow + c + 2, xv.z * alpha);
    atomicAdd(orow + c + 3, xv.w * alpha);
  }
}

// ---------------- bias (+ optional ELU) ----------------
__global__ void k_bias_act(float* __restrict__ out, const float* __restrict__ b,
                           int HF, int do_elu) {
  long long i = (long long)blockIdx.x * blockDim.x + threadIdx.x;
  if (i >= (long long)N_NODES * HF) return;
  int c = (int)(i % HF);
  float v = out[i] + b[c];
  if (do_elu) v = v > 0.f ? v : expm1f(v);
  out[i] = v;
}

// ---------------- final MLP: elu(h@lin1) @ lin2 -> sigmoid ----------------
__global__ void k_final(const float* __restrict__ h, const float* __restrict__ w1,
                        const float* __restrict__ w2, float* __restrict__ out) {
  int n = blockIdx.x * blockDim.x + threadIdx.x;
  if (n >= N_NODES) return;
  const float* row = h + (size_t)n * 128;
  float acc[8] = {0.f, 0.f, 0.f, 0.f, 0.f, 0.f, 0.f, 0.f};
  for (int c = 0; c < 128; ++c) {
    float v = row[c];
#pragma unroll
    for (int j = 0; j < 8; ++j) acc[j] += v * w1[c * 8 + j];
  }
  float z = 0.f;
#pragma unroll
  for (int j = 0; j < 8; ++j) {
    float y = acc[j] > 0.f ? acc[j] : expm1f(acc[j]);
    z += y * w2[j];
  }
  out[n] = 1.f / (1.f + __expf(-z));
}

extern "C" void kernel_launch(void* const* d_in, const int* in_sizes, int n_in,
                              void* d_out, int out_size, void* d_ws, size_t ws_size,
                              hipStream_t stream) {
  const float* x    = (const float*)d_in[0];
  const int*   ei   = (const int*)d_in[1];
  const float* W1   = (const float*)d_in[2];
  const float* a1s  = (const float*)d_in[3];
  const float* a1d  = (const float*)d_in[4];
  const float* b1   = (const float*)d_in[5];
  const float* Wm   = (const float*)d_in[6];
  const float* ams  = (const float*)d_in[7];
  const float* amd_ = (const float*)d_in[8];
  const float* bm   = (const float*)d_in[9];
  const float* W2   = (const float*)d_in[10];
  const float* a2s  = (const float*)d_in[11];
  const float* a2d  = (const float*)d_in[12];
  const float* b2   = (const float*)d_in[13];
  const float* lin1 = (const float*)d_in[14];
  const float* lin2 = (const float*)d_in[15];

  char* ws = (char*)d_ws;
  size_t off = 0;
  auto alloc = [&](size_t bytes) -> void* {
    void* p = ws + off;
    off += (bytes + 255) & ~(size_t)255;
    return p;
  };
  float* P0 = (float*)alloc((size_t)N_NODES * 256 * 4);
  float* P1 = (float*)alloc((size_t)N_NODES * 256 * 4);
  float* P2 = (float*)alloc((size_t)N_NODES * 256 * 4);
  unsigned short* Abf = (unsigned short*)alloc((size_t)N_NODES * 288 * 2);
  unsigned short* Bbf = (unsigned short*)alloc((size_t)288 * 256 * 2);
  float*    als   = (float*)alloc((size_t)N_NODES * 8 * 4);
  float*    ald   = (float*)alloc((size_t)N_NODES * 8 * 4);
  unsigned* smax  = (unsigned*)alloc((size_t)N_NODES * 8 * 4);
  float*    denom = (float*)alloc((size_t)N_NODES * 8 * 4);
  float*    ebuf  = (float*)alloc((size_t)ET_TOTAL * 8 * 4);

  struct L { const float *in, *W, *as, *ad, *b; int K, Kp, Nout, F, elu; float *xh, *out; };
  L ls[4] = {
    { x,  W1,         a1s,       a1d,        b1,       262, 288, 256, 32, 1, P0, P1 },
    { P1, Wm,         ams,       amd_,       bm,       256, 256, 256, 32, 1, P2, P0 },
    { P0, Wm + 65536, ams + 256, amd_ + 256, bm + 256, 256, 256, 256, 32, 1, P1, P2 },
    { P2, W2,         a2s,       a2d,        b2,       256, 256, 128, 16, 0, P0, P1 },
  };

  for (int li = 0; li < 4; ++li) {
    const L& l = ls[li];
    hipMemsetAsync(l.out, 0, (size_t)N_NODES * l.Nout * 4, stream);
    hipMemsetAsync(smax, 0, (size_t)N_NODES * 8 * 4, stream);
    hipMemsetAsync(denom, 0, (size_t)N_NODES * 8 * 4, stream);

    long long na = (long long)N_NODES * l.Kp;
    k_convA<<<(unsigned)((na + 255) / 256), 256, 0, stream>>>(l.in, Abf, l.K, l.Kp);
    int nb = l.Nout * l.Kp;
    k_convB<<<(nb + 255) / 256, 256, 0, stream>>>(l.W, Bbf, l.K, l.Kp, l.Nout);

    int T = (N_NODES / 16) * (l.Nout / 16);
    k_gemm_wmma<<<(T + 3) / 4, 128, 0, stream>>>(Abf, Bbf, l.xh, l.Kp, l.Nout);

    int natt = N_NODES * 8;
    k_node_att<<<(natt + 255) / 256, 256, 0, stream>>>(l.xh, l.as, l.ad, als, ald, l.F);

    long long ne = (long long)ET_TOTAL * 8;
    k_edge_logit<<<(unsigned)((ne + 255) / 256), 256, 0, stream>>>(ei, als, ald, ebuf, smax);
    k_edge_exp<<<(unsigned)((ne + 255) / 256), 256, 0, stream>>>(ei, smax, ebuf, denom);

    k_aggregate<<<(ET_TOTAL + 7) / 8, 256, 0, stream>>>(ei, l.xh, ebuf, denom, l.out,
                                                        l.Nout, l.F);

    long long nba = (long long)N_NODES * l.Nout;
    k_bias_act<<<(unsigned)((nba + 255) / 256), 256, 0, stream>>>(l.out, l.b, l.Nout, l.elu);
  }

  k_final<<<(N_NODES + 127) / 128, 128, 0, stream>>>(P1, lin1, lin2, (float*)d_out);
}
